// Better_Transformer_10874857194110
// MI455X (gfx1250) — matrix operand order repaired and checked by reading, not compiled, not run on hardware
//
#include <hip/hip_runtime.h>

#define IN_SIZE 4096
#define N_PART  32
#define INT_DIM 128
#define TILE_B  128
#define PITCH   136   // bf16 elems per LDS row: 272B pitch -> 16B aligned, conflict-free

typedef __attribute__((ext_vector_type(16))) __bf16 v16bf;
typedef __attribute__((ext_vector_type(8)))  float  v8f;

struct B256 { uint4 lo, hi; };

// f32 -> bf16 with round-to-nearest-even (bit manipulation; avoids relying on
// scalar __bf16 conversion codegen)
__device__ __forceinline__ unsigned f2bf(float f) {
  unsigned u = __builtin_bit_cast(unsigned, f);
  return (u + 0x7fffu + ((u >> 16) & 1u)) >> 16;
}

__global__ __launch_bounds__(256)
void supact_blockdiag_kernel(const float* __restrict__ x,
                             const float* __restrict__ w,
                             const float* __restrict__ bias,
                             const float* __restrict__ gain_p,
                             const float* __restrict__ nb_p,
                             const float* __restrict__ gamma,
                             const float* __restrict__ beta,
                             float* __restrict__ out) {
  extern __shared__ unsigned short smem[];
  unsigned short* As = smem;                   // [128][PITCH], bf16 of (x*gain+nb)
  unsigned short* Ws = smem + TILE_B * PITCH;  // [128][PITCH], bf16 of W_p transposed: Ws[e][d]

  const int t = threadIdx.x;
  const int p = blockIdx.y;
  const int rowBase = blockIdx.x * TILE_B;

  const float gain = gain_p[0];
  const float nb   = nb_p[0];

  // ---- stage X tile: rows rowBase..+128, cols p*128..+128 (affine norm + bf16) ----
  const float* xt = x + (size_t)rowBase * IN_SIZE + p * INT_DIM;
  #pragma unroll
  for (int i = 0; i < 16; ++i) {
    int q = i * 256 + t;          // 0..4095
    int r = q >> 5;               // tile row 0..127
    int c = (q & 31) * 4;         // tile col
    float4 v = *(const float4*)(xt + (size_t)r * IN_SIZE + c);
    uint2 pk;
    pk.x = f2bf(v.x * gain + nb) | (f2bf(v.y * gain + nb) << 16);
    pk.y = f2bf(v.z * gain + nb) | (f2bf(v.w * gain + nb) << 16);
    *(uint2*)(As + r * PITCH + c) = pk;
  }

  // ---- stage W_p transposed: Ws[e][d] = W[p][d][e], via 4x4 register micro-transpose ----
  const float* wp = w + (size_t)p * INT_DIM * INT_DIM;
  #pragma unroll
  for (int i = 0; i < 4; ++i) {
    int q  = i * 256 + t;         // 0..1023 -> 32x32 grid of 4x4 blocks
    int d0 = (q >> 5) * 4;
    int e0 = (q & 31) * 4;
    float4 r0 = *(const float4*)(wp + (d0 + 0) * INT_DIM + e0);
    float4 r1 = *(const float4*)(wp + (d0 + 1) * INT_DIM + e0);
    float4 r2 = *(const float4*)(wp + (d0 + 2) * INT_DIM + e0);
    float4 r3 = *(const float4*)(wp + (d0 + 3) * INT_DIM + e0);
    uint2 c0, c1, c2, c3;
    c0.x = f2bf(r0.x) | (f2bf(r1.x) << 16); c0.y = f2bf(r2.x) | (f2bf(r3.x) << 16);
    c1.x = f2bf(r0.y) | (f2bf(r1.y) << 16); c1.y = f2bf(r2.y) | (f2bf(r3.y) << 16);
    c2.x = f2bf(r0.z) | (f2bf(r1.z) << 16); c2.y = f2bf(r2.z) | (f2bf(r3.z) << 16);
    c3.x = f2bf(r0.w) | (f2bf(r1.w) << 16); c3.y = f2bf(r2.w) | (f2bf(r3.w) << 16);
    *(uint2*)(Ws + (e0 + 0) * PITCH + d0) = c0;
    *(uint2*)(Ws + (e0 + 1) * PITCH + d0) = c1;
    *(uint2*)(Ws + (e0 + 2) * PITCH + d0) = c2;
    *(uint2*)(Ws + (e0 + 3) * PITCH + d0) = c3;
  }

  __syncthreads();

  const int wave = t >> 5;
  const int lane = t & 31;
  const int half = lane >> 4;   // lane group (0: lanes 0-15, 1: lanes 16-31)
  const int ml   = lane & 15;
  const int wr   = wave >> 1;   // 0..3 -> output rows wr*32 .. +32
  const int wc   = wave & 1;    // 0..1 -> output cols wc*64 .. +64

  v8f acc[2][4] = {};

  // ---- main GEMM: K=128 in 4 steps of 32, 2x4 tiles of 16x16 per wave ----
  #pragma unroll
  for (int k0 = 0; k0 < INT_DIM; k0 += 32) {
    v16bf a[2];
    #pragma unroll
    for (int i = 0; i < 2; ++i) {
      // 16-bit A 16x32 layout: lane(m=ml) regs 0-3 hold K = half*8 + 0..7,
      // regs 4-7 hold K = 16 + half*8 + 0..7  -> two contiguous 16B runs
      const unsigned short* ab = As + (wr * 32 + i * 16 + ml) * PITCH + k0 + half * 8;
      B256 tmp;
      tmp.lo = *(const uint4*)(ab);
      tmp.hi = *(const uint4*)(ab + 16);
      a[i] = __builtin_bit_cast(v16bf, tmp);
    }
    #pragma unroll
    for (int j = 0; j < 4; ++j) {
      // 16-bit B 32x16 layout: lane holds col n=ml, regs 0-3 K = half*8 + 0..7,
      // regs 4-7 K = 16 + half*8 + 0..7 (Ws is W^T) -> two contiguous 16B runs
      const unsigned short* bb = Ws + (wc * 64 + j * 16 + ml) * PITCH + k0 + half * 8;
      B256 tmp;
      tmp.lo = *(const uint4*)(bb);
      tmp.hi = *(const uint4*)(bb + 16);
      v16bf bm = __builtin_bit_cast(v16bf, tmp);
      #pragma unroll
      for (int i = 0; i < 2; ++i)
        acc[i][j] = __builtin_amdgcn_wmma_f32_16x16x32_bf16(
            false, a[i], false, bm, (short)0, acc[i][j], false, false);
    }
  }

  // ---- epilogue: bias + supact gate + residual, directly on C/D layout ----
  // C/D 16x16 f32: lane -> col n=ml (half selects +8 row offset), VGPR v -> row v
  #pragma unroll
  for (int j = 0; j < 4; ++j) {
    const int n    = wc * 64 + j * 16 + ml;
    const int gcol = p * INT_DIM + n;
    const float bs = bias[gcol];
    const float gm = gamma[gcol];
    const float bt = beta[gcol];
    const float og = 1.0f - gm;
    #pragma unroll
    for (int i = 0; i < 2; ++i) {
      #pragma unroll
      for (int v = 0; v < 8; ++v) {
        const int m = wr * 32 + i * 16 + v + 8 * half;
        const size_t gidx = (size_t)(rowBase + m) * IN_SIZE + gcol;
        float y = acc[i][j][v] + bs;
        float s = 1.0f / (1.0f + __expf(-bt * y));
        out[gidx] = (gm + s * og) * y + x[gidx];
      }
    }
  }
}

extern "C" void kernel_launch(void* const* d_in, const int* in_sizes, int n_in,
                              void* d_out, int out_size, void* d_ws, size_t ws_size,
                              hipStream_t stream) {
  const float* x     = (const float*)d_in[0];
  const float* w     = (const float*)d_in[1];
  const float* bias  = (const float*)d_in[2];
  const float* gain  = (const float*)d_in[3];
  const float* nb    = (const float*)d_in[4];
  const float* gamma = (const float*)d_in[5];
  const float* beta  = (const float*)d_in[6];
  float* out = (float*)d_out;

  const int B = in_sizes[0] / IN_SIZE;       // 16384
  dim3 grid(B / TILE_B, N_PART, 1);          // (128, 32)
  const size_t lds = (size_t)2 * TILE_B * PITCH * sizeof(unsigned short);  // ~68 KB

  supact_blockdiag_kernel<<<grid, dim3(256, 1, 1), lds, stream>>>(
      x, w, bias, gain, nb, gamma, beta, out);
}
